// LRU_real_77223511982453
// MI455X (gfx1250) — compile-verified, matrix operand order checked by reading, not run on hardware
//
#include <hip/hip_runtime.h>

typedef __bf16 bf16;
typedef __attribute__((ext_vector_type(4)))  bf16  bf16x4;
typedef __attribute__((ext_vector_type(8)))  bf16  bf16x8;
typedef __attribute__((ext_vector_type(16))) bf16  bf16x16;
typedef __attribute__((ext_vector_type(8)))  float v8f;
typedef unsigned int u32x4 __attribute__((ext_vector_type(4)));
typedef int          i32x8 __attribute__((ext_vector_type(8)));
typedef int          i32x4 __attribute__((ext_vector_type(4)));

constexpr int T_SEQ = 16384;
constexpr int DH    = 1024;   // hidden
constexpr int DM    = 1024;   // model
constexpr int KDIM  = 1024;   // contraction dim for both GEMMs

constexpr int BM = 128, BN = 128, BK = 32;
constexpr int LDSPAD = 40;    // 80B row stride: 16B-aligned chunks, conflict-free banks
constexpr int NIT = KDIM / BK;

constexpr int SCAN_L = 128;               // chunk length
constexpr int SCAN_C = T_SEQ / SCAN_L;    // 128 chunks

__device__ __forceinline__ bf16x16 cat8(bf16x8 lo, bf16x8 hi) {
  return __builtin_shufflevector(lo, hi, 0,1,2,3,4,5,6,7,8,9,10,11,12,13,14,15);
}

// ---------------- f32 -> bf16 conversion ----------------
__global__ void cvt_f32_bf16(const float* __restrict__ in, bf16* __restrict__ out, int n) {
  int i = (blockIdx.x * blockDim.x + threadIdx.x) * 4;
  if (i + 3 < n) {
    float4 v = *(const float4*)(in + i);
    bf16x4 o = { (bf16)v.x, (bf16)v.y, (bf16)v.z, (bf16)v.w };
    *(bf16x4*)(out + i) = o;
  }
}

// ---------------- TDM: load one 128x32 bf16 tile into LDS with 16B/row padding -----
// Source: row-major [rows_total][KDIM] bf16, tile starts at gsrc.
// LDS layout produced: 128 rows, 80B stride (40 bf16) == LDSPAD.
__device__ __forceinline__ void tdm_load_tile(unsigned lds_addr, const void* gsrc,
                                              int rows_total) {
  unsigned long long ga = (unsigned long long)(uintptr_t)gsrc;
  u32x4 g0 = {
    1u,                                                 // count=1, user descriptor
    lds_addr,                                           // LDS dest byte address
    (unsigned)ga,                                       // global_addr[31:0]
    (unsigned)((ga >> 32) & 0x01FFFFFFu) | (2u << 30)   // global_addr[56:32] | type=2
  };
  i32x8 g1 = {
    // data_size=1 (2B) | pad_enable | pad_interval=3 (16 DW) | pad_amount=3 (4 DW)
    (1 << 16) | (1 << 20) | (3 << 22) | (3 << 25),
    (int)(((unsigned)KDIM & 0xFFFFu) << 16),                               // dim0[15:0]
    (int)(((unsigned)KDIM >> 16) | (((unsigned)rows_total & 0xFFFFu) << 16)), // dim0[31:16]|dim1[15:0]
    (int)(((unsigned)rows_total >> 16) | ((unsigned)BK << 16)),            // dim1[31:16]|tile_dim0
    BM,                                                                    // tile_dim1 | tile_dim2=0
    KDIM,                                                                  // dim0_stride[31:0]
    0, 0                                                                   // dim0_stride[47:32]|dim1_stride
  };
  i32x4 zz4 = { 0, 0, 0, 0 };
  i32x8 zz8 = { 0, 0, 0, 0, 0, 0, 0, 0 };
  __builtin_amdgcn_tensor_load_to_lds(g0, g1, zz4, zz4, zz8, 0);
}

// ---------------- bf16 WMMA GEMM with TDM double-buffered LDS pipeline ----------------
// Out[M][DM] = A[M][KDIM](bf16) x Bt[N][KDIM](bf16)^T ; EPI: += Dp[n]*X[m][n]
template <bool EPI>
__global__ __launch_bounds__(256) void gemm_bf16_tdm(
    const bf16*  __restrict__ A,    // [a_rows][KDIM] bf16
    const bf16*  __restrict__ Bt,   // [1024][KDIM]  bf16
    float*       __restrict__ Out,  // [M][DM]
    int a_rows,
    const float* __restrict__ Dp,   // [DM]   (EPI only)
    const float* __restrict__ X)    // [M][DM](EPI only)
{
  __shared__ bf16 As[2][BM * LDSPAD];
  __shared__ bf16 Bs[2][BN * LDSPAD];

  const int tid  = threadIdx.x;
  const int lane = tid & 31;
  const int wave = tid >> 5;       // 8 waves
  const int wr   = wave >> 1;      // 0..3 -> M offset wr*32
  const int wc   = wave & 1;       // 0..1 -> N offset wc*64
  const int m0   = blockIdx.y * BM;
  const int n0   = blockIdx.x * BN;
  const int lrow = lane & 15;
  const int kb   = (lane >> 4) << 3;   // 0 or 8 (ISA 16-bit A/B fragment layout)

  const unsigned asb = (unsigned)(uintptr_t)(&As[0][0]);
  const unsigned bsb = (unsigned)(uintptr_t)(&Bs[0][0]);
  constexpr unsigned STAGE = BM * LDSPAD * sizeof(bf16);   // 10240 B

  v8f acc[2][4] = {};

  // Prologue: wave 0 issues TDM loads for stage 0
  if (tid < 32) {
    tdm_load_tile(asb, A  + (size_t)m0 * KDIM, a_rows);
    tdm_load_tile(bsb, Bt + (size_t)n0 * KDIM, 1024);
  }

  for (int it = 0; it < NIT; ++it) {
    const int cur = it & 1;
    if (tid < 32) {
      if (it + 1 < NIT) {
        const int k1 = (it + 1) * BK;
        tdm_load_tile(asb + (cur ^ 1) * STAGE, A  + (size_t)m0 * KDIM + k1, a_rows);
        tdm_load_tile(bsb + (cur ^ 1) * STAGE, Bt + (size_t)n0 * KDIM + k1, 1024);
        __builtin_amdgcn_s_wait_tensorcnt(2);  // stage `cur` complete, prefetch in flight
      } else {
        __builtin_amdgcn_s_wait_tensorcnt(0);
      }
    }
    __syncthreads();   // stage `cur` visible to all waves

    bf16x16 afrag[2], bfrag[4];
    #pragma unroll
    for (int mt = 0; mt < 2; ++mt) {
      int r = wr * 32 + mt * 16 + lrow;
      bf16x8 lo = *(const bf16x8*)&As[cur][r * LDSPAD + kb];
      bf16x8 hi = *(const bf16x8*)&As[cur][r * LDSPAD + kb + 16];
      afrag[mt] = cat8(lo, hi);
    }
    #pragma unroll
    for (int nt = 0; nt < 4; ++nt) {
      int r = wc * 64 + nt * 16 + lrow;
      bf16x8 lo = *(const bf16x8*)&Bs[cur][r * LDSPAD + kb];
      bf16x8 hi = *(const bf16x8*)&Bs[cur][r * LDSPAD + kb + 16];
      bfrag[nt] = cat8(lo, hi);
    }
    #pragma unroll
    for (int mt = 0; mt < 2; ++mt)
      #pragma unroll
      for (int nt = 0; nt < 4; ++nt)
        acc[mt][nt] = __builtin_amdgcn_wmma_f32_16x16x32_bf16(
            false, afrag[mt], false, bfrag[nt], (short)0, acc[mt][nt], false, false);

    __syncthreads();   // all reads of stage `cur` done before it is refilled
  }

  // Epilogue: D layout — VGPR r holds M=r (lanes 0-15) / M=r+8 (lanes 16-31), N=lane%16
  const int mlane = (lane >> 4) << 3;
  const int nlane = lane & 15;
  #pragma unroll
  for (int mt = 0; mt < 2; ++mt)
    #pragma unroll
    for (int nt = 0; nt < 4; ++nt)
      #pragma unroll
      for (int r = 0; r < 8; ++r) {
        int m = m0 + wr * 32 + mt * 16 + r + mlane;
        int n = n0 + wc * 64 + nt * 16 + nlane;
        float v = acc[mt][nt][r];
        if constexpr (EPI) v += Dp[n] * X[(size_t)m * DM + n];
        Out[(size_t)m * DM + n] = v;
      }
}

// ---------------- chunked IIR scan: h_t = lam*h_{t-1} + Bu_t ----------------
__global__ void scan_pass1(const float* __restrict__ Bu, const float* __restrict__ nu,
                           float* __restrict__ carry) {
  int h = blockIdx.y * blockDim.x + threadIdx.x;
  int c = blockIdx.x;
  float lam = nu[h];
  float s = 0.f;
  const float* p = Bu + (size_t)c * SCAN_L * DH + h;
  #pragma unroll 4
  for (int t = 0; t < SCAN_L; ++t) { s = fmaf(lam, s, *p); p += DH; }
  carry[(size_t)c * DH + h] = s;
}

__global__ void scan_pass2(const float* __restrict__ carry, const float* __restrict__ nu,
                           float* __restrict__ H0) {
  int h = blockIdx.x * blockDim.x + threadIdx.x;
  float lam = nu[h];
  float lamL = lam;
  #pragma unroll
  for (int i = 0; i < 7; ++i) lamL *= lamL;   // lam^128
  float s = 0.f;
  for (int c = 0; c < SCAN_C; ++c) {
    H0[(size_t)c * DH + h] = s;
    s = lamL * s + carry[(size_t)c * DH + h];
  }
}

__global__ void scan_pass3(const float* __restrict__ Bu, const float* __restrict__ nu,
                           const float* __restrict__ H0, bf16* __restrict__ Hb) {
  int h = blockIdx.y * blockDim.x + threadIdx.x;
  int c = blockIdx.x;
  float lam = nu[h];
  float s = H0[(size_t)c * DH + h];
  const float* p = Bu + (size_t)c * SCAN_L * DH + h;
  bf16* q = Hb + (size_t)c * SCAN_L * DH + h;
  #pragma unroll 4
  for (int t = 0; t < SCAN_L; ++t) { s = fmaf(lam, s, *p); *q = (bf16)s; p += DH; q += DH; }
}

// ---------------- launch ----------------
extern "C" void kernel_launch(void* const* d_in, const int* in_sizes, int n_in,
                              void* d_out, int out_size, void* d_ws, size_t ws_size,
                              hipStream_t stream) {
  const float* X  = (const float*)d_in[0];   // [T, DM]
  const float* nu = (const float*)d_in[1];   // [DH]
  const float* Bm = (const float*)d_in[2];   // [DH, DM]
  const float* Cm = (const float*)d_in[3];   // [DM, DH]
  const float* Dp = (const float*)d_in[4];   // [DM]
  float* Out = (float*)d_out;

  char* ws = (char*)d_ws;
  size_t off = 0;
  auto take = [&](size_t bytes) -> char* {
    char* p = ws + off;
    off = (off + bytes + 255) & ~(size_t)255;
    return p;
  };
  bf16*  Xb    = (bf16*) take((size_t)T_SEQ * DM * sizeof(bf16));
  bf16*  Bb    = (bf16*) take((size_t)DH * DM * sizeof(bf16));
  bf16*  Cb    = (bf16*) take((size_t)DM * DH * sizeof(bf16));
  float* Bu    = (float*)take((size_t)T_SEQ * DH * sizeof(float));
  bf16*  Hb    = (bf16*) take((size_t)T_SEQ * DH * sizeof(bf16));
  float* carry = (float*)take((size_t)SCAN_C * DH * sizeof(float));
  float* H0    = (float*)take((size_t)SCAN_C * DH * sizeof(float));

  // 1. inputs/weights -> bf16
  cvt_f32_bf16<<<(T_SEQ * DM) / 1024, 256, 0, stream>>>(X, Xb, T_SEQ * DM);
  cvt_f32_bf16<<<(DH * DM) / 1024, 256, 0, stream>>>(Bm, Bb, DH * DM);
  cvt_f32_bf16<<<(DM * DH) / 1024, 256, 0, stream>>>(Cm, Cb, DM * DH);
  // 2. Bu = X @ B^T  (bf16 WMMA, TDM-fed LDS, f32 accumulate)
  gemm_bf16_tdm<false><<<dim3(DH / BN, T_SEQ / BM), 256, 0, stream>>>(
      Xb, Bb, Bu, T_SEQ, nullptr, nullptr);
  // 3. chunked scan (constant-lambda IIR), hidden emitted as bf16
  scan_pass1<<<dim3(SCAN_C, DH / 256), 256, 0, stream>>>(Bu, nu, carry);
  scan_pass2<<<DH / 256, 256, 0, stream>>>(carry, nu, H0);
  scan_pass3<<<dim3(SCAN_C, DH / 256), 256, 0, stream>>>(Bu, nu, H0, Hb);
  // 4. Out = hidden @ C^T + Dp * X  (bf16 WMMA, fused epilogue)
  gemm_bf16_tdm<true><<<dim3(DM / BN, T_SEQ / BM), 256, 0, stream>>>(
      Hb, Cb, Out, T_SEQ, Dp, X);
}
